// Conv_89369679495334
// MI455X (gfx1250) — matrix operand order repaired and checked by reading, compiled, not run on hardware
//
#include <hip/hip_runtime.h>
#include <hip/hip_bf16.h>

typedef __attribute__((ext_vector_type(16))) _Float16 v16h;
typedef __attribute__((ext_vector_type(8)))  _Float16 v8h;
typedef __attribute__((ext_vector_type(8)))  float    v8f;

#define E_EDGES 4096
#define NNODES  256

// ---------------------------------------------------------------------------
// Conversion / weight repack
// ---------------------------------------------------------------------------
__global__ void k_f32_to_f16(const float* __restrict__ in, _Float16* __restrict__ out, int n) {
  int i = blockIdx.x * blockDim.x + threadIdx.x;
  if (i < n) out[i] = (_Float16)in[i];
}

// w: [COUT][CIN][3][3] f32  ->  wf: [COUT][9*CP] f16, K index = tap*CP + ci
template<int CIN, int COUT>
__global__ void k_prep_w(const float* __restrict__ w, _Float16* __restrict__ wf) {
  constexpr int CP = ((CIN + 31) / 32) * 32;
  constexpr int KP = 9 * CP;
  int total = COUT * KP;
  for (int idx = blockIdx.x * blockDim.x + threadIdx.x; idx < total;
       idx += gridDim.x * blockDim.x) {
    int co  = idx / KP;
    int r   = idx - co * KP;
    int tap = r / CP;
    int ci  = r - tap * CP;
    float v = (ci < CIN) ? w[(co * CIN + ci) * 9 + tap] : 0.0f;
    wf[idx] = (_Float16)v;
  }
}

// ---------------------------------------------------------------------------
// Fused 3x3 SAME conv + bias + relu + 2x2 maxpool, implicit GEMM on WMMA.
// Block = 128 threads = 4 waves (2 M-slabs x 2 row-pair groups).
// Each wave: 1 A fragment (16 cout) shared across 2 B fragments (2x16 pixels)
// -> 2 WMMAs per K step, halving global weight traffic per WMMA.
// Workgroup tile: 32 cout x 8x8 conv pixels -> 32 cout x 4x4 pooled pixels.
// Activations between layers are channel-last f16 ([H][W][C], C % 32 == 0) so
// the LDS im2col patch is staged with coalesced 128-bit copies. Layer 1
// (CIN==3) takes the scalar path with zero channel padding to CP=32.
// K loop is software-pipelined (double-buffered fragments) so the WMMAs for
// step s overlap the global/LDS loads of step s+1.
// ---------------------------------------------------------------------------
template<int CIN, int COUT, int HIN, int WIN, bool WRITE_F32>
__global__ __launch_bounds__(128)
void k_conv3x3_relu_pool(const _Float16* __restrict__ act_in,
                         const _Float16* __restrict__ wf,       // [COUT][9*CP]
                         const float*    __restrict__ bias,     // [COUT]
                         _Float16*       __restrict__ act_out,  // [HO][WO][COUT]
                         float*          __restrict__ org_out)  // f32 [COUT][HO*WO]
{
  constexpr int CP = ((CIN + 31) / 32) * 32;
  constexpr int KP = 9 * CP;
  constexpr int HO = HIN / 2, WO = WIN / 2;
  constexpr int NSTEP = 9 * (CP / 32);

  __shared__ __align__(32) _Float16 patch[100 * CP];   // [10x10 pixels][CP channels]
  __shared__ float convbuf[32 * 64];                   // [co_local][8x8]

  const int tid    = threadIdx.x;
  const int tileX0 = blockIdx.x * 8;
  const int tileY0 = blockIdx.y * 8;
  const int co0    = blockIdx.z * 32;

  if constexpr (CIN == 3) {
    // zero channel padding, then scalar stage from channel-major input
    for (int idx = tid; idx < 100 * CP; idx += 128) patch[idx] = (_Float16)0.0f;
    __syncthreads();
    for (int idx = tid; idx < CIN * 100; idx += 128) {
      int ci = idx / 100;
      int p  = idx - ci * 100;
      int py = p / 10, px = p - py * 10;
      int gy = tileY0 + py - 1, gx = tileX0 + px - 1;
      float v = 0.0f;
      if (gy >= 0 && gy < HIN && gx >= 0 && gx < WIN)
        v = (float)act_in[(size_t)ci * HIN * WIN + (size_t)gy * WIN + gx];
      patch[p * CP + ci] = (_Float16)v;
    }
  } else {
    // channel-last input: vectorized 128-bit staging (8 f16 per copy)
    constexpr int CV = CP / 8;
    for (int idx = tid; idx < 100 * CV; idx += 128) {
      int p  = idx / CV, c8 = idx - p * CV;
      int py = p / 10, px = p - py * 10;
      int gy = tileY0 + py - 1, gx = tileX0 + px - 1;
      v8h v = {};
      if (gy >= 0 && gy < HIN && gx >= 0 && gx < WIN)
        v = ((const v8h*)(act_in + ((size_t)gy * WIN + gx) * CIN))[c8];
      ((v8h*)(patch + p * CP))[c8] = v;
    }
  }
  __syncthreads();

  const int wid   = tid >> 5;          // 0..3
  const int lane  = tid & 31;
  const int mwave = wid >> 1;          // 0..1 : which 16-cout slab
  const int nwave = wid & 1;           // 0..1 : which 4-row group
  const int g     = lane >> 4;         // lane half
  const int nn    = lane & 15;
  const int ry0   = nwave * 4 + (nn >> 3);   // B-tile 0: conv rows {4n, 4n+1}
  const int ry1   = ry0 + 2;                 // B-tile 1: conv rows {4n+2, 4n+3}
  const int rx    = nn & 7;                  // conv col 0..7 in tile
  const int co_row = co0 + mwave * 16 + nn;  // A-fragment row (M = lane&15)

  const _Float16* wbase = wf + (size_t)co_row * KP + g * 16;

  // A fragment: lane half g holds K in [k0+16g, +16)
  auto ldA = [&](int s) -> v16h {
    const v8h* wp = (const v8h*)(wbase + s * 32);
    v8h lo = wp[0];
    v8h hi = wp[1];
    v16h a;
#pragma unroll
    for (int i = 0; i < 8; ++i) { a[i] = lo[i]; a[8 + i] = hi[i]; }
    return a;
  };
  // B fragment: lane half g holds K in [k0+16g, +16) — contiguous channels
  auto ldB = [&](int s, int ry) -> v16h {
    int tap = s / (CP / 32);
    int c0  = (s - tap * (CP / 32)) * 32;
    int ky  = tap / 3, kx = tap - ky * 3;
    const _Float16* bsrc = &patch[((ry + ky) * 10 + (rx + kx)) * CP];
    return *(const v16h*)(bsrc + c0 + g * 16);
  };

  v8f acc0 = {}, acc1 = {};
  v16h a_cur  = ldA(0);
  v16h b0_cur = ldB(0, ry0);
  v16h b1_cur = ldB(0, ry1);
#pragma unroll
  for (int s = 0; s < NSTEP; ++s) {
    v16h a_nxt = a_cur, b0_nxt = b0_cur, b1_nxt = b1_cur;
    if (s + 1 < NSTEP) {
      a_nxt  = ldA(s + 1);
      b0_nxt = ldB(s + 1, ry0);
      b1_nxt = ldB(s + 1, ry1);
    }
    acc0 = __builtin_amdgcn_wmma_f32_16x16x32_f16(false, a_cur, false, b0_cur,
                                                  (short)0, acc0, false, false);
    acc1 = __builtin_amdgcn_wmma_f32_16x16x32_f16(false, a_cur, false, b1_cur,
                                                  (short)0, acc1, false, false);
    a_cur = a_nxt; b0_cur = b0_nxt; b1_cur = b1_nxt;
  }

  // Epilogue: bias + relu into LDS conv buffer. D layout: reg r -> M = r + 8g
  const float* bp = bias + co0 + mwave * 16;
#pragma unroll
  for (int r = 0; r < 8; ++r) {
    int m = r + 8 * g;
    float b  = bp[m];
    float v0 = fmaxf(acc0[r] + b, 0.0f);
    float v1 = fmaxf(acc1[r] + b, 0.0f);
    convbuf[(mwave * 16 + m) * 64 + ry0 * 8 + rx] = v0;
    convbuf[(mwave * 16 + m) * 64 + ry1 * 8 + rx] = v1;
  }
  __syncthreads();

  // 2x2 max pool -> channel-last f16 (and channel-major f32 for final layer)
  for (int o = tid; o < 32 * 16; o += 128) {
    int co_l = o >> 4;
    int q  = o & 15;
    int py = q >> 2, px = q & 3;
    const float* cb = &convbuf[co_l * 64 + (py * 2) * 8 + px * 2];
    float v = fmaxf(fmaxf(cb[0], cb[1]), fmaxf(cb[8], cb[9]));
    int oy = blockIdx.y * 4 + py;
    int ox = blockIdx.x * 4 + px;
    act_out[((size_t)oy * WO + ox) * COUT + (co0 + co_l)] = (_Float16)v;
    if (WRITE_F32)
      org_out[(size_t)(co0 + co_l) * HO * WO + (size_t)oy * WO + ox] = v;
  }
}

// ---------------------------------------------------------------------------
// Graph-side kernels (tiny, f32)
// ---------------------------------------------------------------------------
__global__ void k_zero(float* p, int n) {
  int i = blockIdx.x * blockDim.x + threadIdx.x;
  if (i < n) p[i] = 0.0f;
}

__global__ void k_scatter_feat(const int* __restrict__ src, const int* __restrict__ dst,
                               const float* __restrict__ org,
                               float* __restrict__ tot, float* __restrict__ cnt) {
  int e = blockIdx.x;
  int f = threadIdx.x;
  int s = src[e], d = dst[e];
  atomicAdd(&tot[d * NNODES + f], org[s * NNODES + f]);
  if (f == 0) atomicAdd(&cnt[d], 1.0f);
}

__global__ void k_scatter_adj(const int* __restrict__ src, const int* __restrict__ dst,
                              float* __restrict__ adj) {
  int e = blockIdx.x * blockDim.x + threadIdx.x;
  if (e < E_EDGES) atomicAdd(&adj[src[e] * NNODES + dst[e]], 1.0f);
}

__global__ void k_mean(const float* __restrict__ tot, const float* __restrict__ cnt,
                       float* __restrict__ agg) {
  int idx = blockIdx.x * blockDim.x + threadIdx.x;   // 65536
  int i = idx >> 8;
  float c = cnt[i];
  agg[idx] = c > 0.0f ? tot[idx] / fmaxf(c, 1.0f) : 0.0f;
}

// x1 = agg@s1_wl + org@s1_wr + s1_b (256x2); s = agg@s2_wl + org@s2_wr + s2_b (256x128)
__global__ void k_sage12(const float* __restrict__ agg, const float* __restrict__ org,
                         const float* s1wl, const float* s1wr, const float* s1b,
                         const float* s2wl, const float* s2wr, const float* s2b,
                         float* __restrict__ x1, float* __restrict__ s) {
  int i = blockIdx.x, j = threadIdx.x;
  const float* ai = agg + i * NNODES;
  const float* oi = org + i * NNODES;
  if (j < 128) {
    float a = s2b[j];
    for (int c = 0; c < NNODES; ++c)
      a += ai[c] * s2wl[c * 128 + j] + oi[c] * s2wr[c * 128 + j];
    s[i * 128 + j] = a;
  } else if (j < 130) {
    int t = j - 128;
    float a = s1b[t];
    for (int c = 0; c < NNODES; ++c)
      a += ai[c] * s1wl[c * 2 + t] + oi[c] * s1wr[c * 2 + t];
    x1[i * 2 + t] = a;
  }
}

__global__ void k_softmax_ent(const float* __restrict__ s, float* __restrict__ ssoft,
                              float* __restrict__ ent_acc) {
  __shared__ float red[128];
  int i = blockIdx.x, j = threadIdx.x;
  float v = s[i * 128 + j];
  red[j] = v; __syncthreads();
  for (int off = 64; off > 0; off >>= 1) {
    if (j < off) red[j] = fmaxf(red[j], red[j + off]);
    __syncthreads();
  }
  float mx = red[0]; __syncthreads();
  float e = expf(v - mx);
  red[j] = e; __syncthreads();
  for (int off = 64; off > 0; off >>= 1) {
    if (j < off) red[j] += red[j + off];
    __syncthreads();
  }
  float sum = red[0]; __syncthreads();
  float p = e / sum;
  ssoft[i * 128 + j] = p;
  red[j] = -p * logf(p + 1e-15f); __syncthreads();
  for (int off = 64; off > 0; off >>= 1) {
    if (j < off) red[j] += red[j + off];
    __syncthreads();
  }
  if (j == 0) atomicAdd(ent_acc, red[0] * (1.0f / 256.0f));
}

__global__ void k_adj_s(const float* __restrict__ adj, const float* __restrict__ ssoft,
                        float* __restrict__ tmp) {
  int i = blockIdx.x, j = threadIdx.x;
  float a = 0.0f;
  for (int k = 0; k < NNODES; ++k) a += adj[i * NNODES + k] * ssoft[k * 128 + j];
  tmp[i * 128 + j] = a;
}

__global__ void k_adjpool(const float* __restrict__ ssoft, const float* __restrict__ tmp,
                          float* __restrict__ adj_pool) {
  int p = blockIdx.x, q = threadIdx.x;
  float a = 0.0f;
  for (int i = 0; i < NNODES; ++i) a += ssoft[i * 128 + p] * tmp[i * 128 + q];
  adj_pool[p * 128 + q] = a;
}

__global__ void k_xpool(const float* __restrict__ ssoft, const float* __restrict__ x1,
                        float* __restrict__ x_pool) {
  int idx = threadIdx.x;            // 256
  int p = idx >> 1, t = idx & 1;
  float a = 0.0f;
  for (int i = 0; i < NNODES; ++i) a += ssoft[i * 128 + p] * x1[i * 2 + t];
  x_pool[idx] = a;
}

__global__ void k_link(const float* __restrict__ adj, const float* __restrict__ ssoft,
                       float* __restrict__ link_acc) {
  __shared__ float red[256];
  int i = blockIdx.x, j = threadIdx.x;
  float d = 0.0f;
  for (int c = 0; c < 128; ++c) d += ssoft[i * 128 + c] * ssoft[j * 128 + c];
  d = adj[i * NNODES + j] - d;
  red[j] = d * d; __syncthreads();
  for (int off = 128; off > 0; off >>= 1) {
    if (j < off) red[j] += red[j + off];
    __syncthreads();
  }
  if (j == 0) atomicAdd(link_acc, red[0]);
}

__global__ void k_nodes(const float* __restrict__ x_pool, float* __restrict__ nodes,
                        float* __restrict__ out_nodes) {
  int idx = threadIdx.x;
  float v = tanhf(x_pool[idx]);
  nodes[idx] = v;
  out_nodes[idx] = v;
}

__global__ void k_bin(const float* __restrict__ adj_pool, float* __restrict__ ebin,
                      float* __restrict__ out_bin) {
  __shared__ float red[128];
  int p = blockIdx.x, q = threadIdx.x;
  float v = adj_pool[p * 128 + q];
  red[q] = v; __syncthreads();
  for (int off = 64; off > 0; off >>= 1) {
    if (q < off) red[q] = fmaxf(red[q], red[q + off]);
    __syncthreads();
  }
  float b = (v == red[0]) ? 1.0f : 0.0f;
  ebin[p * 128 + q] = b;
  out_bin[p * 128 + q] = b;
}

__global__ void k_out(const float* __restrict__ ebin, const float* __restrict__ nodes,
                      const float* wl, const float* wr, const float* sb,
                      float* __restrict__ out) {
  __shared__ float r0[128], r1[128], r2[128];
  int q = blockIdx.x, p = threadIdx.x;
  float eb = ebin[p * 128 + q];
  r0[p] = eb;
  r1[p] = eb * nodes[p * 2 + 0];
  r2[p] = eb * nodes[p * 2 + 1];
  __syncthreads();
  for (int off = 64; off > 0; off >>= 1) {
    if (p < off) { r0[p] += r0[p + off]; r1[p] += r1[p + off]; r2[p] += r2[p + off]; }
    __syncthreads();
  }
  if (p < 2) {
    int t = p;
    float ind = r0[0];
    float g0 = ind > 0.0f ? r1[0] / fmaxf(ind, 1.0f) : 0.0f;
    float g1 = ind > 0.0f ? r2[0] / fmaxf(ind, 1.0f) : 0.0f;
    float n0 = nodes[q * 2 + 0], n1 = nodes[q * 2 + 1];
    out[q * 2 + t] = g0 * wl[t] + g1 * wl[2 + t] + sb[t] + n0 * wr[t] + n1 * wr[2 + t];
  }
}

__global__ void k_final(const float* __restrict__ scal, float* __restrict__ out) {
  out[256] = sqrtf(scal[0]) * (1.0f / 65536.0f);   // link_loss
  out[257] = scal[1];                              // ent_loss
}

// ---------------------------------------------------------------------------
extern "C" void kernel_launch(void* const* d_in, const int* in_sizes, int n_in,
                              void* d_out, int out_size, void* d_ws, size_t ws_size,
                              hipStream_t stream) {
  (void)in_sizes; (void)n_in; (void)out_size; (void)ws_size;
  const float* inputs = (const float*)d_in[0];
  const int*   ei     = (const int*)d_in[1];
  const int*   src    = ei;
  const int*   dst    = ei + E_EDGES;
  const float* w1 = (const float*)d_in[2];  const float* cb1 = (const float*)d_in[3];
  const float* w2 = (const float*)d_in[4];  const float* cb2 = (const float*)d_in[5];
  const float* w3 = (const float*)d_in[6];  const float* cb3 = (const float*)d_in[7];
  const float* w4 = (const float*)d_in[8];  const float* cb4 = (const float*)d_in[9];
  const float* w5 = (const float*)d_in[10]; const float* cb5 = (const float*)d_in[11];
  const float* s1wl = (const float*)d_in[12];
  const float* s1wr = (const float*)d_in[13];
  const float* s1b  = (const float*)d_in[14];
  const float* s2wl = (const float*)d_in[15];
  const float* s2wr = (const float*)d_in[16];
  const float* s2b  = (const float*)d_in[17];
  const float* s3wl = (const float*)d_in[18];
  const float* s3wr = (const float*)d_in[19];
  const float* s3b  = (const float*)d_in[20];
  float* out = (float*)d_out;

  char* ws = (char*)d_ws;
  size_t off = 0;
  auto alloc = [&](size_t bytes) -> void* {
    void* p = ws + off;
    off += (bytes + 255) & ~(size_t)255;
    return p;
  };

  _Float16* a0  = (_Float16*)alloc((size_t)3 * 512 * 512 * 2);    // [C][H][W]
  _Float16* a1  = (_Float16*)alloc((size_t)256 * 256 * 32 * 2);   // [H][W][C]
  _Float16* a2  = (_Float16*)alloc((size_t)128 * 128 * 64 * 2);
  _Float16* a3  = (_Float16*)alloc((size_t)64 * 64 * 128 * 2);
  _Float16* a4  = (_Float16*)alloc((size_t)32 * 32 * 256 * 2);
  _Float16* a5  = (_Float16*)alloc((size_t)16 * 16 * 256 * 2);
  float*    org = (float*)alloc((size_t)256 * 256 * 4);           // [C][pixel]
  _Float16* wf1 = (_Float16*)alloc((size_t)32 * 9 * 32 * 2);
  _Float16* wf2 = (_Float16*)alloc((size_t)64 * 9 * 32 * 2);
  _Float16* wf3 = (_Float16*)alloc((size_t)128 * 9 * 64 * 2);
  _Float16* wf4 = (_Float16*)alloc((size_t)256 * 9 * 128 * 2);
  _Float16* wf5 = (_Float16*)alloc((size_t)256 * 9 * 256 * 2);
  float* tot      = (float*)alloc(65536 * 4);
  float* cnt      = (float*)alloc(256 * 4);
  float* agg      = (float*)alloc(65536 * 4);
  float* x1       = (float*)alloc(512 * 4);
  float* sbuf     = (float*)alloc(32768 * 4);
  float* ssoft    = (float*)alloc(32768 * 4);
  float* adj      = (float*)alloc(65536 * 4);
  float* tmp      = (float*)alloc(32768 * 4);
  float* adj_pool = (float*)alloc(16384 * 4);
  float* x_pool   = (float*)alloc(256 * 4);
  float* nodes    = (float*)alloc(256 * 4);
  float* ebin     = (float*)alloc(16384 * 4);
  float* scal     = (float*)alloc(2 * 4);      // [0]=link acc, [1]=ent acc

  // --- conversions / weight repack (recomputed every call: deterministic) ---
  { int n = 3 * 512 * 512;
    k_f32_to_f16<<<(n + 255) / 256, 256, 0, stream>>>(inputs, a0, n); }
  k_prep_w<3, 32>  <<<64, 256, 0, stream>>>(w1, wf1);
  k_prep_w<32, 64> <<<64, 256, 0, stream>>>(w2, wf2);
  k_prep_w<64, 128><<<128, 256, 0, stream>>>(w3, wf3);
  k_prep_w<128, 256><<<256, 256, 0, stream>>>(w4, wf4);
  k_prep_w<256, 256><<<512, 256, 0, stream>>>(w5, wf5);

  // --- fused conv+relu+pool chain (WMMA implicit GEMM) ---
  k_conv3x3_relu_pool<3, 32, 512, 512, false>
      <<<dim3(64, 64, 1), 128, 0, stream>>>(a0, wf1, cb1, a1, nullptr);
  k_conv3x3_relu_pool<32, 64, 256, 256, false>
      <<<dim3(32, 32, 2), 128, 0, stream>>>(a1, wf2, cb2, a2, nullptr);
  k_conv3x3_relu_pool<64, 128, 128, 128, false>
      <<<dim3(16, 16, 4), 128, 0, stream>>>(a2, wf3, cb3, a3, nullptr);
  k_conv3x3_relu_pool<128, 256, 64, 64, false>
      <<<dim3(8, 8, 8), 128, 0, stream>>>(a3, wf4, cb4, a4, nullptr);
  k_conv3x3_relu_pool<256, 256, 32, 32, true>
      <<<dim3(4, 4, 8), 128, 0, stream>>>(a4, wf5, cb5, a5, org);

  // --- graph part ---
  k_zero<<<(65536 + 255) / 256, 256, 0, stream>>>(tot, 65536);
  k_zero<<<1, 256, 0, stream>>>(cnt, 256);
  k_zero<<<(65536 + 255) / 256, 256, 0, stream>>>(adj, 65536);
  k_zero<<<1, 256, 0, stream>>>(scal, 2);

  k_scatter_feat<<<E_EDGES, 256, 0, stream>>>(src, dst, org, tot, cnt);
  k_scatter_adj<<<16, 256, 0, stream>>>(src, dst, adj);
  k_mean<<<256, 256, 0, stream>>>(tot, cnt, agg);
  k_sage12<<<256, 256, 0, stream>>>(agg, org, s1wl, s1wr, s1b, s2wl, s2wr, s2b, x1, sbuf);
  k_softmax_ent<<<256, 128, 0, stream>>>(sbuf, ssoft, scal + 1);
  k_adj_s<<<256, 128, 0, stream>>>(adj, ssoft, tmp);
  k_adjpool<<<128, 128, 0, stream>>>(ssoft, tmp, adj_pool);
  k_xpool<<<1, 256, 0, stream>>>(ssoft, x1, x_pool);
  k_link<<<256, 256, 0, stream>>>(adj, ssoft, scal);
  k_nodes<<<1, 256, 0, stream>>>(x_pool, nodes, out + 258);
  k_bin<<<128, 128, 0, stream>>>(adj_pool, ebin, out + 514);
  k_out<<<128, 128, 0, stream>>>(ebin, nodes, s3wl, s3wr, s3b, out);
  k_final<<<1, 1, 0, stream>>>(scal, out);
}
